// GraphSAGEEncoder_11029476016739
// MI455X (gfx1250) — matrix operand order repaired and checked by reading, not compiled, not run on hardware
//
#include <hip/hip_runtime.h>
#include <hip/hip_bf16.h>

typedef __bf16 bf16;
typedef __attribute__((ext_vector_type(16))) __bf16 v16bf;
typedef __attribute__((ext_vector_type(8)))  __bf16 v8bf;
typedef __attribute__((ext_vector_type(8)))  float  v8f;

#define FH    64
#define FE    8
#define KTOT  160           // 64 (agg x) + 8 (agg e) + 64 (x) + 24 pad
#define KC    5             // K chunks of 32 for bf16 WMMA
#define ZSTR  168           // z-tile row stride in bf16 (16B aligned, bank-spread)
#define OSTR  68            // out-tile row stride in f32 (bank-spread)
#define WVB   5376          // per-wave LDS scratch bytes (>= 16*ZSTR*2 and >= 16*OSTR*4)

static __device__ inline v16bf cat8(v8bf lo, v8bf hi) {
  return __builtin_shufflevector(lo, hi, 0,1,2,3,4,5,6,7,8,9,10,11,12,13,14,15);
}
static __device__ inline v16bf load16(const bf16* p) {
  const v8bf* q = (const v8bf*)p;
  return cat8(q[0], q[1]);
}

// ---------------- zero workspace region ----------------
__global__ void gnn_zero_kernel(float* __restrict__ p, long long n) {
  long long i = (long long)blockIdx.x * blockDim.x + threadIdx.x;
  if (i < n) p[i] = 0.0f;
}

// ---------------- per-edge aggregation (gather + scatter-add) ----------------
// 8 threads per edge; thread p handles x[src] features [8p, 8p+8); p==0 also
// does the 8 edge_attr features and the degree count. x/acc arrays live in L2.
__global__ __launch_bounds__(256) void gnn_scatter_kernel(
    const int* __restrict__ eidx, const float* __restrict__ x,
    const float* __restrict__ ea, float* __restrict__ accA,
    float* __restrict__ accB, float* __restrict__ cnt, long long E) {
  long long gid = (long long)blockIdx.x * 256 + threadIdx.x;
  long long e = gid >> 3;
  int p = (int)(gid & 7);
  if (e >= E) return;
  int src = eidx[e];
  int dst = eidx[E + e];
  const float4* xs = (const float4*)(x + (long long)src * FH + p * 8);
  float4 a = xs[0], b = xs[1];
  float* dA = accA + (long long)dst * FH + p * 8;
  atomicAdd(dA + 0, a.x); atomicAdd(dA + 1, a.y);
  atomicAdd(dA + 2, a.z); atomicAdd(dA + 3, a.w);
  atomicAdd(dA + 4, b.x); atomicAdd(dA + 5, b.y);
  atomicAdd(dA + 6, b.z); atomicAdd(dA + 7, b.w);
  if (p == 0) {
    const float4* es = (const float4*)(ea + e * FE);
    float4 c = es[0], d = es[1];
    float* dB = accB + (long long)dst * FE;
    atomicAdd(dB + 0, c.x); atomicAdd(dB + 1, c.y);
    atomicAdd(dB + 2, c.z); atomicAdd(dB + 3, c.w);
    atomicAdd(dB + 4, d.x); atomicAdd(dB + 5, d.y);
    atomicAdd(dB + 6, d.z); atomicAdd(dB + 7, d.w);
    atomicAdd(cnt + dst, 1.0f);
  }
}

// ---------------- build combined bf16 weight [K=160][64]: (k,n) = Wcat[n][k] ----------------
__global__ void gnn_prep_w_kernel(const float* __restrict__ Wn,
                                  const float* __restrict__ Wr,
                                  bf16* __restrict__ out) {
  int idx = blockIdx.x * 256 + threadIdx.x;
  if (idx >= KTOT * FH) return;
  int n = idx & 63, k = idx >> 6;
  float v = 0.0f;
  if (k < 72)       v = Wn[n * 72 + k];         // [x_src | edge_attr] part
  else if (k < 136) v = Wr[n * 64 + (k - 72)];  // root part
  out[idx] = (bf16)v;
}

__global__ void gnn_prep_wp_kernel(const float* __restrict__ Wp, bf16* __restrict__ out) {
  int idx = blockIdx.x * 256 + threadIdx.x;
  if (idx >= 64 * 64) return;
  int n = idx & 63, k = idx >> 6;
  out[idx] = (bf16)Wp[n * 64 + k];
}

// ---------------- fused node update: (scaled-agg | x) @ Wcat.T + bias, relu, layernorm ----------------
// block = 256 = 8 waves; each wave owns a 16-node tile; WMMA bf16 16x16x32, K=160.
__global__ __launch_bounds__(256) void gnn_node_update_kernel(
    const float* __restrict__ xin, const float* __restrict__ accA,
    const float* __restrict__ accB, const float* __restrict__ cnt,
    const bf16* __restrict__ wprep, const float* __restrict__ bn,
    const float* __restrict__ br, const float* __restrict__ gam,
    const float* __restrict__ bet, float* __restrict__ xout, int N) {
  __shared__ __attribute__((aligned(16))) bf16 wLds[KC * 4 * 32 * 16];  // 20 KB, pre-swizzled B
  __shared__ __attribute__((aligned(16))) char wsc[8 * WVB];            // 43 KB, per-wave z/out tiles
  int tid = threadIdx.x;

  // stage weights in per-lane WMMA B layout: elem j of lane L (ntile nt, chunk kc)
  // holds Wprep[k][n] with k = kc*32 + (L>=16)*8 + (j&7) + (j>=8)*16, n = nt*16 + (L&15)
  for (int idx = tid; idx < KC * 4 * 32 * 16; idx += 256) {
    int j = idx & 15, lane = (idx >> 4) & 31, nt = (idx >> 9) & 3, kc = idx >> 11;
    int k = kc * 32 + ((lane >> 4) << 3) + (j & 7) + ((j >> 3) << 4);
    int n = nt * 16 + (lane & 15);
    wLds[idx] = wprep[k * FH + n];
  }
  __syncthreads();

  int wave = tid >> 5, lane = tid & 31;
  int tileBase = (blockIdx.x * 8 + wave) * 16;
  bf16*  zT   = (bf16*)(wsc + wave * WVB);
  float* outT = (float*)(wsc + wave * WVB);  // overlaid; separated by barriers

  // build z tile: [16 nodes][160] bf16 = [accA/cnt | accB/cnt | x | 0]
  for (int r = 0; r < 16; ++r) {
    int node = tileBase + r;
    bool ok = node < N;
    float c = ok ? cnt[node] : 0.0f;
    float inv = c > 0.0f ? 1.0f / c : 0.0f;
    for (int k = lane; k < KTOT; k += 32) {
      float v;
      if (k < 64)       v = (ok ? accA[node * FH + k] : 0.0f) * inv;
      else if (k < 72)  v = (ok ? accB[node * FE + (k - 64)] : 0.0f) * inv;
      else if (k < 136) v =  ok ? xin[node * FH + (k - 72)] : 0.0f;
      else              v = 0.0f;
      zT[r * ZSTR + k] = (bf16)v;
    }
  }
  __syncthreads();

  v8f c0 = {}, c1 = {}, c2 = {}, c3 = {};
  int row = lane & 15;
  int kb  = (lane >> 4) << 3;  // A layout: lanes<16 hold K {0-7,16-23}, lanes>=16 {8-15,24-31}
  for (int kc = 0; kc < KC; ++kc) {
    const bf16* zp = zT + row * ZSTR + kc * 32 + kb;
    v16bf a = cat8(*(const v8bf*)zp, *(const v8bf*)(zp + 16));
    v16bf b0 = load16(wLds + ((kc * 4 + 0) * 32 + lane) * 16);
    v16bf b1 = load16(wLds + ((kc * 4 + 1) * 32 + lane) * 16);
    v16bf b2 = load16(wLds + ((kc * 4 + 2) * 32 + lane) * 16);
    v16bf b3 = load16(wLds + ((kc * 4 + 3) * 32 + lane) * 16);
    c0 = __builtin_amdgcn_wmma_f32_16x16x32_bf16(false, a, false, b0, (short)0, c0, false, false);
    c1 = __builtin_amdgcn_wmma_f32_16x16x32_bf16(false, a, false, b1, (short)0, c1, false, false);
    c2 = __builtin_amdgcn_wmma_f32_16x16x32_bf16(false, a, false, b2, (short)0, c2, false, false);
    c3 = __builtin_amdgcn_wmma_f32_16x16x32_bf16(false, a, false, b3, (short)0, c3, false, false);
  }
  __syncthreads();  // all z-tile reads done before overlaying with outT

  int hi = lane >> 4, colb = lane & 15;
#pragma unroll
  for (int v = 0; v < 8; ++v) {
    int r2 = (v + 8 * hi) * OSTR;
    outT[r2 +  0 + colb] = c0[v];
    outT[r2 + 16 + colb] = c1[v];
    outT[r2 + 32 + colb] = c2[v];
    outT[r2 + 48 + colb] = c3[v];
  }
  __syncthreads();

  // epilogue: +br + (cnt>0)*bn, relu, layernorm over 64 feats; one lane per node row
  if (lane < 16) {
    int node = tileBase + lane;
    if (node < N) {
      float c = cnt[node];
      float fl = c > 0.0f ? 1.0f : 0.0f;
      float sum = 0.0f, sq = 0.0f;
      for (int j = 0; j < FH; ++j) {
        float v = outT[lane * OSTR + j] + br[j] + fl * bn[j];
        v = v > 0.0f ? v : 0.0f;
        outT[lane * OSTR + j] = v;
        sum += v; sq += v * v;
      }
      float mu = sum * (1.0f / FH);
      float var = sq * (1.0f / FH) - mu * mu;
      var = var > 0.0f ? var : 0.0f;
      float rinv = rsqrtf(var + 1e-5f);
      for (int j = 0; j < FH; ++j) {
        float v = (outT[lane * OSTR + j] - mu) * rinv * gam[j] + bet[j];
        xout[node * FH + j] = v;
      }
    }
  }
}

// ---------------- attention scores: s = tanh(x2 @ Wp.T + bp) @ Ws + bs ----------------
#define ZSTR2 72
#define WVB2  4352
__global__ __launch_bounds__(256) void gnn_pool_score_kernel(
    const float* __restrict__ x2, const bf16* __restrict__ wpp,
    const float* __restrict__ bp, const float* __restrict__ Ws,
    const float* __restrict__ bs, float* __restrict__ sout, int N) {
  __shared__ __attribute__((aligned(16))) bf16 wLds[2 * 4 * 32 * 16];  // 8 KB
  __shared__ __attribute__((aligned(16))) char wsc[8 * WVB2];          // 34 KB
  int tid = threadIdx.x;
  for (int idx = tid; idx < 2 * 4 * 32 * 16; idx += 256) {
    int j = idx & 15, lane = (idx >> 4) & 31, nt = (idx >> 9) & 3, kc = idx >> 11;
    int k = kc * 32 + ((lane >> 4) << 3) + (j & 7) + ((j >> 3) << 4);
    int n = nt * 16 + (lane & 15);
    wLds[idx] = wpp[k * FH + n];
  }
  __syncthreads();

  int wave = tid >> 5, lane = tid & 31;
  int tileBase = (blockIdx.x * 8 + wave) * 16;
  bf16*  zT   = (bf16*)(wsc + wave * WVB2);
  float* outT = (float*)(wsc + wave * WVB2);

  for (int r = 0; r < 16; ++r) {
    int node = tileBase + r;
    bool ok = node < N;
    for (int k = lane; k < FH; k += 32)
      zT[r * ZSTR2 + k] = (bf16)(ok ? x2[node * FH + k] : 0.0f);
  }
  __syncthreads();

  v8f c0 = {}, c1 = {}, c2 = {}, c3 = {};
  int row = lane & 15;
  int kb = (lane >> 4) << 3;
  for (int kc = 0; kc < 2; ++kc) {
    const bf16* zp = zT + row * ZSTR2 + kc * 32 + kb;
    v16bf a = cat8(*(const v8bf*)zp, *(const v8bf*)(zp + 16));
    v16bf b0 = load16(wLds + ((kc * 4 + 0) * 32 + lane) * 16);
    v16bf b1 = load16(wLds + ((kc * 4 + 1) * 32 + lane) * 16);
    v16bf b2 = load16(wLds + ((kc * 4 + 2) * 32 + lane) * 16);
    v16bf b3 = load16(wLds + ((kc * 4 + 3) * 32 + lane) * 16);
    c0 = __builtin_amdgcn_wmma_f32_16x16x32_bf16(false, a, false, b0, (short)0, c0, false, false);
    c1 = __builtin_amdgcn_wmma_f32_16x16x32_bf16(false, a, false, b1, (short)0, c1, false, false);
    c2 = __builtin_amdgcn_wmma_f32_16x16x32_bf16(false, a, false, b2, (short)0, c2, false, false);
    c3 = __builtin_amdgcn_wmma_f32_16x16x32_bf16(false, a, false, b3, (short)0, c3, false, false);
  }
  __syncthreads();
  int hi = lane >> 4, colb = lane & 15;
#pragma unroll
  for (int v = 0; v < 8; ++v) {
    int r2 = (v + 8 * hi) * OSTR;
    outT[r2 +  0 + colb] = c0[v];
    outT[r2 + 16 + colb] = c1[v];
    outT[r2 + 32 + colb] = c2[v];
    outT[r2 + 48 + colb] = c3[v];
  }
  __syncthreads();
  if (lane < 16) {
    int node = tileBase + lane;
    if (node < N) {
      float s = bs[0];
      for (int j = 0; j < FH; ++j) {
        float h = tanhf(outT[lane * OSTR + j] + bp[j]);
        s += h * Ws[j];
      }
      sout[node] = s;
    }
  }
}

// ---------------- pooling softmax pieces ----------------
__global__ void gnn_pool_init_kernel(unsigned* __restrict__ smax, float* __restrict__ den,
                                     float* __restrict__ out, int G) {
  int i = blockIdx.x * 256 + threadIdx.x;
  if (i < G) { smax[i] = 0x007FFFFFu; den[i] = 0.0f; }  // encode(-inf)
  if (i < G * FH) out[i] = 0.0f;
}

__global__ void gnn_segmax_kernel(const float* __restrict__ s, const int* __restrict__ batch,
                                  unsigned* __restrict__ smax, int N) {
  int i = blockIdx.x * 256 + threadIdx.x;
  if (i >= N) return;
  unsigned u = __float_as_uint(s[i]);
  u = (u & 0x80000000u) ? ~u : (u | 0x80000000u);  // order-preserving map
  atomicMax(&smax[batch[i]], u);
}

__global__ void gnn_expw_kernel(const float* __restrict__ s, const int* __restrict__ batch,
                                const unsigned* __restrict__ smax, float* __restrict__ w,
                                float* __restrict__ den, int N) {
  int i = blockIdx.x * 256 + threadIdx.x;
  if (i >= N) return;
  unsigned u = smax[batch[i]];
  float m = (u & 0x80000000u) ? __uint_as_float(u & 0x7FFFFFFFu) : __uint_as_float(~u);
  float e = expf(s[i] - m);
  w[i] = e;
  atomicAdd(&den[batch[i]], e);
}

// batch is sorted: accumulate per-feature partial sums across a node chunk,
// flushing with one atomicAdd per graph boundary per feature.
#define PCHUNK 256
__global__ __launch_bounds__(64) void gnn_pooled_kernel(
    const float* __restrict__ x2, const float* __restrict__ w,
    const float* __restrict__ den, const int* __restrict__ batch,
    float* __restrict__ out, int N) {
  int f = threadIdx.x;  // 0..63
  int start = blockIdx.x * PCHUNK;
  if (start >= N) return;
  int end = start + PCHUNK; if (end > N) end = N;
  int curg = batch[start];
  float acc = 0.0f;
  for (int n = start; n < end; ++n) {
    int g = batch[n];
    if (g != curg) { atomicAdd(&out[curg * FH + f], acc); acc = 0.0f; curg = g; }
    float wn = w[n] / den[g];
    acc += x2[(long long)n * FH + f] * wn;
  }
  atomicAdd(&out[curg * FH + f], acc);
}

// ---------------- host ----------------
extern "C" void kernel_launch(void* const* d_in, const int* in_sizes, int n_in,
                              void* d_out, int out_size, void* d_ws, size_t ws_size,
                              hipStream_t stream) {
  const float* x    = (const float*)d_in[0];
  const int*   eidx = (const int*)d_in[1];
  const float* ea   = (const float*)d_in[2];
  const int*   batch = (const int*)d_in[3];
  const float* Wn1 = (const float*)d_in[5];  const float* bn1 = (const float*)d_in[6];
  const float* Wr1 = (const float*)d_in[7];  const float* br1 = (const float*)d_in[8];
  const float* g1  = (const float*)d_in[9];  const float* b1  = (const float*)d_in[10];
  const float* Wn2 = (const float*)d_in[11]; const float* bn2 = (const float*)d_in[12];
  const float* Wr2 = (const float*)d_in[13]; const float* br2 = (const float*)d_in[14];
  const float* g2  = (const float*)d_in[15]; const float* b2  = (const float*)d_in[16];
  const float* Wp  = (const float*)d_in[17]; const float* bp  = (const float*)d_in[18];
  const float* Ws  = (const float*)d_in[19]; const float* bs  = (const float*)d_in[20];

  const int N = in_sizes[0] / FH;
  const long long E = (long long)in_sizes[1] / 2;
  const int G = out_size / FH;
  float* outp = (float*)d_out;

  // workspace layout
  float* accA = (float*)d_ws;                       // N*64
  float* accB = accA + (size_t)N * FH;              // N*8
  float* cnt  = accB + (size_t)N * FE;              // N
  float* x1   = cnt  + (size_t)N;                   // N*64
  float* x2   = x1   + (size_t)N * FH;              // N*64
  float* sb   = x2   + (size_t)N * FH;              // N
  float* wb   = sb   + (size_t)N;                   // N
  float* den  = wb   + (size_t)N;                   // G
  unsigned* smax = (unsigned*)(den + G);            // G
  bf16* wprep  = (bf16*)(smax + G);                 // 160*64
  bf16* wpprep = wprep + KTOT * FH;                 // 64*64

  const long long nzero = (long long)N * (FH + FE + 1);
  dim3 blk256(256), blk64(64);
  int zb  = (int)((nzero + 255) / 256);
  int sbk = (int)((E * 8 + 255) / 256);
  int nbk = (N + 127) / 128;
  int ebk = (N + 255) / 256;
  int pbk = (N + PCHUNK - 1) / PCHUNK;

  // ---- layer 1 ----
  gnn_zero_kernel<<<zb, blk256, 0, stream>>>(accA, nzero);
  gnn_scatter_kernel<<<sbk, blk256, 0, stream>>>(eidx, x, ea, accA, accB, cnt, E);
  gnn_prep_w_kernel<<<(KTOT * FH + 255) / 256, blk256, 0, stream>>>(Wn1, Wr1, wprep);
  gnn_node_update_kernel<<<nbk, blk256, 0, stream>>>(x, accA, accB, cnt, wprep,
                                                     bn1, br1, g1, b1, x1, N);
  // ---- layer 2 ----
  gnn_zero_kernel<<<zb, blk256, 0, stream>>>(accA, nzero);
  gnn_scatter_kernel<<<sbk, blk256, 0, stream>>>(eidx, x1, ea, accA, accB, cnt, E);
  gnn_prep_w_kernel<<<(KTOT * FH + 255) / 256, blk256, 0, stream>>>(Wn2, Wr2, wprep);
  gnn_node_update_kernel<<<nbk, blk256, 0, stream>>>(x1, accA, accB, cnt, wprep,
                                                     bn2, br2, g2, b2, x2, N);
  // ---- attention pooling ----
  gnn_prep_wp_kernel<<<(64 * 64 + 255) / 256, blk256, 0, stream>>>(Wp, wpprep);
  gnn_pool_score_kernel<<<nbk, blk256, 0, stream>>>(x2, wpprep, bp, Ws, bs, sb, N);
  gnn_pool_init_kernel<<<(G * FH + 255) / 256, blk256, 0, stream>>>(smax, den, outp, G);
  gnn_segmax_kernel<<<ebk, blk256, 0, stream>>>(sb, batch, smax, N);
  gnn_expw_kernel<<<ebk, blk256, 0, stream>>>(sb, batch, smax, wb, den, N);
  gnn_pooled_kernel<<<pbk, blk64, 0, stream>>>(x2, wb, den, batch, outp, N);
}
